// NRC_MLP_61753039782448
// MI455X (gfx1250) — matrix operand order repaired and checked by reading, not compiled, not run on hardware
//
#include <hip/hip_runtime.h>

// ---------------------------------------------------------------------------
// Fully-fused NRC-style MLP for gfx1250 (MI455X, wave32, WMMA).
//   encode(64 feats) -> 6 x [64x64] ReLU layers -> [3x64] head -> *(alpha+beta)
// WMMA roles: A = weights (M = out features), B = activations (N = samples),
// so D fragments are feature-contiguous per sample -> packed ReLU + b128 LDS
// writeback. Each wave owns 32 samples (2 N-tiles sharing A fragments).
// ---------------------------------------------------------------------------

#define B_TOTAL       (1u << 20)             // 1,048,576 samples
#define NWAVES        8                      // waves per workgroup
#define TPB           (NWAVES * 32)          // 256 threads
#define SPW           32                     // samples per wave
#define ROWS_PER_WG   (NWAVES * SPW)         // 256 samples per tile
#define NTILES        (B_TOTAL / ROWS_PER_WG) // 4096
#define GRID          1024                   // each WG loops 4 tiles

#define WROW 72   // LDS row stride (halves) for weights  (odd 16-byte multiple)
#define AROW 72   // LDS row stride (halves) for activations
#define LDS_HALVES (6 * 64 * WROW + 16 * WROW + NWAVES * SPW * AROW)

typedef _Float16 h8   __attribute__((ext_vector_type(8)));
typedef _Float16 h16  __attribute__((ext_vector_type(16)));
typedef float    f8   __attribute__((ext_vector_type(8)));

__device__ __forceinline__ h16 cat16(h8 lo, h8 hi) {
    return __builtin_shufflevector(lo, hi, 0,1,2,3,4,5,6,7,8,9,10,11,12,13,14,15);
}

// A fragment (16x32 f16, M=weight rows): lane holds row, halves
// {kb..kb+7} and {kb+16..kb+23} within a 32-wide K window.
__device__ __forceinline__ h16 load_a_frag(const _Float16* kwin, int kb) {
    h8 lo = *(const h8*)(kwin + kb);        // ds_load_b128
    h8 hi = *(const h8*)(kwin + kb + 16);   // ds_load_b128
    return cat16(lo, hi);
}

// B fragment (32x16 f16, N=samples): lane holds K values (features)
// {kb..kb+7} and {kb+16..kb+23} of one sample row.
__device__ __forceinline__ h16 load_b_frag(const _Float16* ptr) {
    h8 lo = *(const h8*)(ptr);              // ds_load_b128
    h8 hi = *(const h8*)(ptr + 16);         // ds_load_b128
    return cat16(lo, hi);
}

__device__ __forceinline__ f8 wmma16(h16 a, h16 b, f8 c) {
    return __builtin_amdgcn_wmma_f32_16x16x32_f16(
        /*neg_a=*/false, a, /*neg_b=*/false, b,
        /*c_mod=*/(short)0, c, /*reuse_a=*/false, /*reuse_b=*/false);
}

// f32x8 -> f16x8 with packed ReLU (v_cvt + v_pk_max_num_f16)
__device__ __forceinline__ h8 relu8(f8 c) {
    h8 h = __builtin_convertvector(c, h8);
    h8 z = {};
    return __builtin_elementwise_max(h, z);
}

// unit-square direction parameterization + one-blob (k=4) -> 8 features
template <int BASE>
__device__ __forceinline__ void sph_oneblob(float x, float y, float z, h8* enc) {
    const float PI = 3.14159265358979323846f;
    float inv = 1.0f / (sqrtf(x * x + y * y + z * z) + 1e-8f);
    float nx = x * inv, ny = y * inv, nz = z * inv;
    float u = atan2f(ny, nx) * (0.5f / PI) + 0.5f;
    float cz = fminf(fmaxf(nz, -1.0f + 1e-6f), 1.0f - 1e-6f);
    float v = acosf(cz) * (1.0f / PI);
#pragma unroll
    for (int i = 0; i < 4; ++i) {
        float c = (i + 0.5f) * 0.25f;
        float tu = (u - c) * 4.0f;
        float tv = (v - c) * 4.0f;
        enc[(BASE + i) >> 3][(BASE + i) & 7]         = (_Float16)__expf(-0.5f * tu * tu);
        enc[(BASE + 4 + i) >> 3][(BASE + 4 + i) & 7] = (_Float16)__expf(-0.5f * tv * tv);
    }
}

__global__ __launch_bounds__(TPB) void nrc_fused_mlp(
    const float* __restrict__ gp,  const float* __restrict__ gwi,
    const float* __restrict__ gn,  const float* __restrict__ gal,
    const float* __restrict__ gbe, const float* __restrict__ gr,
    const float* __restrict__ W0, const float* __restrict__ W1,
    const float* __restrict__ W2, const float* __restrict__ W3,
    const float* __restrict__ W4, const float* __restrict__ W5,
    const float* __restrict__ W6, float* __restrict__ out)
{
    extern __shared__ __align__(16) _Float16 lds[];
    _Float16* wlds = lds;                       // 6 layers: [64][WROW] each
    _Float16* wfin = wlds + 6 * 64 * WROW;      // head padded to [16][WROW]
    _Float16* acts = wfin + 16 * WROW;          // per wave: [SPW][AROW]

    const int tid  = threadIdx.x;
    const int wave = tid >> 5;
    const int lane = tid & 31;
    const int m    = lane & 15;
    const int hi   = (lane >> 4) & 1;

    // ---- Stage all weights into LDS as f16 (once per workgroup) ----
    const float* Wg[6] = {W0, W1, W2, W3, W4, W5};
#pragma unroll
    for (int l = 0; l < 6; ++l) {
        _Float16* dst = wlds + l * 64 * WROW;
        for (int i = tid; i < 64 * 64; i += TPB) {
            int n = i >> 6, k = i & 63;
            dst[n * WROW + k] = (_Float16)Wg[l][i];
        }
    }
    for (int i = tid; i < 16 * 64; i += TPB) {
        int n = i >> 6, k = i & 63;
        wfin[n * WROW + k] = (n < 3) ? (_Float16)W6[n * 64 + k] : (_Float16)0.0f;
    }
    __syncthreads();

    _Float16* areg = acts + wave * (SPW * AROW);   // this wave's 32 samples
    const int kbA = hi * 8;    // A-fragment K base within 32-wide window
    const int kbB = hi * 8;    // B-fragment K base within 32-wide window

    for (unsigned tile = blockIdx.x; tile < NTILES; tile += gridDim.x) {
        __syncthreads();  // prior iteration's act reads before encode overwrite

        // ---- Encode: every thread builds one sample's 64 features ----
        {
            const int s = tile * ROWS_PER_WG + tid;
            _Float16* a = acts + (tid >> 5) * (SPW * AROW) + (tid & 31) * AROW;
            const float PI = 3.14159265358979323846f;

            h8 enc[8];
#define ENC(i, v) enc[(i) >> 3][(i) & 7] = (_Float16)(v)
            float pv[3] = {gp[3 * s], gp[3 * s + 1], gp[3 * s + 2]};
#pragma unroll
            for (int d = 0; d < 3; ++d) {
                float f = PI * pv[d];
#pragma unroll
                for (int j = 0; j < 6; ++j) {      // per dim: 6 sin then 6 cos
                    ENC(d * 12 + j,     __sinf(f));
                    ENC(d * 12 + 6 + j, __cosf(f));
                    f *= 2.0f;
                }
            }
            sph_oneblob<36>(gwi[3 * s], gwi[3 * s + 1], gwi[3 * s + 2], enc);
            sph_oneblob<44>(gn [3 * s], gn [3 * s + 1], gn [3 * s + 2], enc);
            {
                float x = 1.0f - __expf(-gr[s]);
#pragma unroll
                for (int i = 0; i < 4; ++i) {
                    float c = (i + 0.5f) * 0.25f;
                    float t = (x - c) * 4.0f;
                    ENC(52 + i, __expf(-0.5f * t * t));
                }
            }
            ENC(56, gal[3 * s]); ENC(57, gal[3 * s + 1]); ENC(58, gal[3 * s + 2]);
            ENC(59, gbe[3 * s]); ENC(60, gbe[3 * s + 1]); ENC(61, gbe[3 * s + 2]);
            ENC(62, 1.0f); ENC(63, 1.0f);
#undef ENC
#pragma unroll
            for (int i = 0; i < 8; ++i)
                *(h8*)(a + 8 * i) = enc[i];        // 8x ds_store_b128
        }
        __syncthreads();

        const _Float16* r0 = areg + m * AROW;          // sample tile 0 row
        const _Float16* r1 = areg + (16 + m) * AROW;   // sample tile 1 row

        // ---- 6 hidden layers: act = ReLU(W @ act) via WMMA ----
#pragma unroll 1
        for (int l = 0; l < 6; ++l) {
            const _Float16* wb = wlds + l * 64 * WROW;
            // B fragments (activations) for both sample tiles, K=0..63
            h16 b00 = load_b_frag(r0 + kbB);
            h16 b01 = load_b_frag(r0 + 32 + kbB);
            h16 b10 = load_b_frag(r1 + kbB);
            h16 b11 = load_b_frag(r1 + 32 + kbB);
#pragma unroll
            for (int mt = 0; mt < 4; ++mt) {           // output-feature tiles
                const _Float16* wr = wb + (mt * 16 + m) * WROW;
                h16 a0 = load_a_frag(wr, kbA);         // K = 0..31
                h16 a1 = load_a_frag(wr + 32, kbA);    // K = 32..63
                f8 c0 = {}, c1 = {};
                c0 = wmma16(a0, b00, c0);
                c0 = wmma16(a1, b01, c0);
                c1 = wmma16(a0, b10, c1);
                c1 = wmma16(a1, b11, c1);
                // D lane holds 8 contiguous features of its sample:
                //   features mt*16 + hi*8 .. +7 of sample row m (per tile)
                *(h8*)(areg + m * AROW        + mt * 16 + hi * 8) = relu8(c0);
                *(h8*)(areg + (16 + m) * AROW + mt * 16 + hi * 8) = relu8(c1);
            }
        }

        // ---- Head: channels = M rows of one padded 16x16 tile ----
        {
            h16 b00 = load_b_frag(r0 + kbB);
            h16 b01 = load_b_frag(r0 + 32 + kbB);
            h16 b10 = load_b_frag(r1 + kbB);
            h16 b11 = load_b_frag(r1 + 32 + kbB);
            const _Float16* wr = wfin + m * WROW;
            h16 a0 = load_a_frag(wr, kbA);
            h16 a1 = load_a_frag(wr + 32, kbA);
            f8 c0 = {}, c1 = {};
            c0 = wmma16(a0, b00, c0);
            c0 = wmma16(a1, b01, c0);
            c1 = wmma16(a0, b10, c1);
            c1 = wmma16(a1, b11, c1);

            if (hi == 0) {   // lanes 0..15 hold channels 0..2 (M=0..2)
                const int s0 = tile * ROWS_PER_WG + wave * SPW + m;
                const int s1 = s0 + 16;
#pragma unroll
                for (int c = 0; c < 3; ++c) {
                    out[s0 * 3 + c] = c0[c] * (gal[s0 * 3 + c] + gbe[s0 * 3 + c]);
                    out[s1 * 3 + c] = c1[c] * (gal[s1 * 3 + c] + gbe[s1 * 3 + c]);
                }
            }
        }
    }
}

extern "C" void kernel_launch(void* const* d_in, const int* in_sizes, int n_in,
                              void* d_out, int out_size, void* d_ws, size_t ws_size,
                              hipStream_t stream) {
    const float* p  = (const float*)d_in[0];
    const float* wi = (const float*)d_in[1];
    const float* nr = (const float*)d_in[2];
    const float* al = (const float*)d_in[3];
    const float* be = (const float*)d_in[4];
    const float* r  = (const float*)d_in[5];
    const float* W0 = (const float*)d_in[6];
    const float* W1 = (const float*)d_in[7];
    const float* W2 = (const float*)d_in[8];
    const float* W3 = (const float*)d_in[9];
    const float* W4 = (const float*)d_in[10];
    const float* W5 = (const float*)d_in[11];
    const float* W6 = (const float*)d_in[12];
    float* out = (float*)d_out;

    const size_t shmem = (size_t)LDS_HALVES * sizeof(_Float16); // 94464 B
    nrc_fused_mlp<<<GRID, TPB, shmem, stream>>>(
        p, wi, nr, al, be, r, W0, W1, W2, W3, W4, W5, W6, out);
}
